// PseudoEntropy_22445499089270
// MI455X (gfx1250) — matrix operand order repaired and checked by reading, not compiled, not run on hardware
//
#include <hip/hip_runtime.h>
#include <cfloat>

#define N_PTS 8192
#define DIMS  256
#define KNN   8
// bf16 elements per LDS row: 264*2 = 528 B = 132 dwords, 132 % 64 = 4 -> the
// 16 row-strided fragment lanes hit distinct LDS banks. 528 is 16B-aligned.
#define LDA   264

typedef __attribute__((ext_vector_type(4)))  __bf16 v4bf;
typedef __attribute__((ext_vector_type(8)))  __bf16 v8bf;
typedef __attribute__((ext_vector_type(16))) __bf16 v16bf;
typedef __attribute__((ext_vector_type(8)))  float  v8f;

__device__ __forceinline__ v8f wmma_bf16(v16bf a, v16bf b, v8f c) {
  // D = A(16x32) * B(32x16) + C, fp32 accumulate (CDNA5 V_WMMA_F32_16X16X32_BF16)
  return __builtin_amdgcn_wmma_f32_16x16x32_bf16(false, a, false, b, (short)0, c, false, false);
}

// concat two 8-element bf16 LDS chunks into one v16bf operand
__device__ __forceinline__ v16bf ld_frag(const __bf16* p0, const __bf16* p1) {
  v8bf a = *(const v8bf*)p0;
  v8bf b = *(const v8bf*)p1;
  return __builtin_shufflevector(a, b, 0, 1, 2, 3, 4, 5, 6, 7, 8, 9, 10, 11, 12, 13, 14, 15);
}

// split fp32 quad into bf16 hi + bf16 lo residual, store both to LDS
__device__ __forceinline__ void split_store(float4 v, __bf16* hrow, __bf16* lrow) {
  __bf16 h0 = (__bf16)v.x, h1 = (__bf16)v.y, h2 = (__bf16)v.z, h3 = (__bf16)v.w;
  v4bf hv = {h0, h1, h2, h3};
  v4bf lv = {(__bf16)(v.x - (float)h0), (__bf16)(v.y - (float)h1),
             (__bf16)(v.z - (float)h2), (__bf16)(v.w - (float)h3)};
  *(v4bf*)hrow = hv;
  *(v4bf*)lrow = lv;
}

// ---------------------------------------------------------------------------
// Kernel 0: zero accumulators (workspace is poisoned by the harness)
// ---------------------------------------------------------------------------
__global__ void init_kernel(double* gsum, float* colsum, float* colsq) {
  int t = threadIdx.x;
  if (t == 0) *gsum = 0.0;
  colsum[t] = 0.0f;
  colsq[t]  = 0.0f;
}

// ---------------------------------------------------------------------------
// Kernel 1: per-row squared norms, exact fp32 (one wave32 per row)
// ---------------------------------------------------------------------------
__global__ __launch_bounds__(256) void rownorm_kernel(const float* __restrict__ e,
                                                      float* __restrict__ sqn) {
  int row  = blockIdx.x * 8 + (threadIdx.x >> 5);
  int lane = threadIdx.x & 31;
  const float* p = e + (size_t)row * DIMS;
  float acc = 0.0f;
#pragma unroll
  for (int j = 0; j < 8; ++j) { float v = p[lane + 32 * j]; acc = fmaf(v, v, acc); }
  acc += __shfl_xor(acc, 1);
  acc += __shfl_xor(acc, 2);
  acc += __shfl_xor(acc, 4);
  acc += __shfl_xor(acc, 8);
  acc += __shfl_xor(acc, 16);
  if (lane == 0) sqn[row] = acc;
}

// ---------------------------------------------------------------------------
// Kernel 2: per-column sum / sum-of-squares (ddof=1 variance), coalesced rows
// ---------------------------------------------------------------------------
__global__ __launch_bounds__(256) void colstats_kernel(const float* __restrict__ e,
                                                       float* __restrict__ colsum,
                                                       float* __restrict__ colsq) {
  int c  = threadIdx.x;
  int r0 = blockIdx.x * 256;
  float s = 0.0f, s2 = 0.0f;
  for (int r = 0; r < 256; ++r) {
    float v = e[(size_t)(r0 + r) * DIMS + c];
    s += v;
    s2 = fmaf(v, v, s2);
  }
  atomicAdd(&colsum[c], s);
  atomicAdd(&colsq[c], s2);
}

// ---------------------------------------------------------------------------
// Kernel 3: fused  G = e*e^T  via split-bf16 WMMA  +  streaming top-8.
//   G ~= Ah*Bh^T + Ah*Bl^T + Al*Bh^T   (lo*lo term ~2^-18, dropped)
// Block: 256 threads (8 waves), 64 rows. Waves: wm 0..3 (16-row strip),
// wn 0..1 (32-col strip -> two 16x16 tiles). 128 column tiles of 64.
// ---------------------------------------------------------------------------
__global__ __launch_bounds__(256) void gemm_topk_kernel(const float* __restrict__ e,
                                                        const float* __restrict__ sqn,
                                                        double* __restrict__ gsum) {
  __shared__ __bf16 Ah[64 * LDA];   // 33 KB each, 132 KB total
  __shared__ __bf16 Al[64 * LDA];
  __shared__ __bf16 Bh[64 * LDA];
  __shared__ __bf16 Bl[64 * LDA];

  const int tid  = threadIdx.x;
  const int lane = tid & 31;
  const int wid  = tid >> 5;
  const int wm   = wid & 3;       // row wave 0..3
  const int wn   = wid >> 2;      // col wave 0..1
  const int hi   = lane >> 4;     // lane half
  const int l15  = lane & 15;
  const int rowBase = blockIdx.x * 64;

  // ---- stage + split A rows (64 x 256 fp32 -> bf16 hi/lo) once ----
#pragma unroll
  for (int it = 0; it < 16; ++it) {
    int linear = it * 256 + tid;
    int j = linear >> 6;          // row 0..63
    int q = linear & 63;          // float4 index within row
    float4 v = *(const float4*)&e[(size_t)(rowBase + j) * DIMS + q * 4];
    split_store(v, &Ah[j * LDA + q * 4], &Al[j * LDA + q * 4]);
  }

  // per-lane row sq-norms for the 8 C-slots (lane half selects M=i vs M=i+8)
  float nrow[8];
#pragma unroll
  for (int i = 0; i < 8; ++i)
    nrow[i] = sqn[rowBase + wm * 16 + i + 8 * hi];

  // per-lane sorted-ascending top-8 lists, one per C row-slot
  float s[8][8];
#pragma unroll
  for (int i = 0; i < 8; ++i)
#pragma unroll
    for (int j = 0; j < 8; ++j) s[i][j] = FLT_MAX;

  // Fragment bases. A (16-bit, 16x32): lane m -> K in [16h, 16h+16) contiguous.
  // B (32x16): lane n -> K in [16h, 16h+16) contiguous.
  const int arow = (wm * 16 + l15) * LDA + 16 * hi;
  const __bf16* AhF = &Ah[arow];
  const __bf16* AlF = &Al[arow];
  const int bcol0 = (wn * 32 + l15) * LDA + 16 * hi;
  const int bcol1 = (wn * 32 + 16 + l15) * LDA + 16 * hi;
  const __bf16* BhF0 = &Bh[bcol0];
  const __bf16* BlF0 = &Bl[bcol0];
  const __bf16* BhF1 = &Bh[bcol1];
  const __bf16* BlF1 = &Bl[bcol1];

  for (int ct = 0; ct < N_PTS / 64; ++ct) {
    const int colBase = ct * 64;
    __syncthreads();  // previous compute done before overwriting B tiles
    // ---- stage + split B tile (64 cols x 256 K) ----
#pragma unroll
    for (int it = 0; it < 16; ++it) {
      int linear = it * 256 + tid;
      int j = linear >> 6;
      int q = linear & 63;
      float4 v = *(const float4*)&e[(size_t)(colBase + j) * DIMS + q * 4];
      split_store(v, &Bh[j * LDA + q * 4], &Bl[j * LDA + q * 4]);
    }
    __syncthreads();

    // prefetch next tile into L2 (global_prefetch_b8)
    if (ct + 1 < N_PTS / 64) {
      const float* nb = &e[(size_t)(colBase + 64) * DIMS];
      __builtin_prefetch(nb + tid * 64, 0, 1);
      __builtin_prefetch(nb + tid * 64 + 32, 0, 1);
    }

    // ---- K loop: 8 steps of K=32, 3 split terms x 2 N-tiles = 6 WMMA ----
    v8f c0 = {};
    v8f c1 = {};
#pragma unroll
    for (int k = 0; k < DIMS; k += 32) {
      v16bf ah  = ld_frag(AhF + k,  AhF + k + 8);
      v16bf al  = ld_frag(AlF + k,  AlF + k + 8);
      v16bf bh0 = ld_frag(BhF0 + k, BhF0 + k + 8);
      v16bf bl0 = ld_frag(BlF0 + k, BlF0 + k + 8);
      v16bf bh1 = ld_frag(BhF1 + k, BhF1 + k + 8);
      v16bf bl1 = ld_frag(BlF1 + k, BlF1 + k + 8);
      c0 = wmma_bf16(ah, bh0, c0);
      c1 = wmma_bf16(ah, bh1, c1);
      c0 = wmma_bf16(ah, bl0, c0);
      c1 = wmma_bf16(ah, bl1, c1);
      c0 = wmma_bf16(al, bh0, c0);
      c1 = wmma_bf16(al, bh1, c1);
    }

    // ---- squared distances + gated sorted-insert (branchless bubble) ----
    float nc0 = sqn[colBase + wn * 32 + l15];
    float nc1 = sqn[colBase + wn * 32 + 16 + l15];
#pragma unroll
    for (int i = 0; i < 8; ++i) {
      float sq0 = fmaxf(fmaf(-2.0f, c0[i], nrow[i] + nc0), 0.0f);
      if (__any(sq0 < s[i][7])) {
        float cc = sq0;
#pragma unroll
        for (int j = 0; j < 8; ++j) {
          float lo = fminf(s[i][j], cc); cc = fmaxf(s[i][j], cc); s[i][j] = lo;
        }
      }
      float sq1 = fmaxf(fmaf(-2.0f, c1[i], nrow[i] + nc1), 0.0f);
      if (__any(sq1 < s[i][7])) {
        float cc = sq1;
#pragma unroll
        for (int j = 0; j < 8; ++j) {
          float lo = fminf(s[i][j], cc); cc = fmaxf(s[i][j], cc); s[i][j] = lo;
        }
      }
    }
  }

  __syncthreads();
  // ---- cross-lane pop-merge: per row-slot, pop 8 smallest across the 16
  // column-lanes of this wave half via shfl_xor min-reductions (wave32) ----
  float* mbuf = (float*)Ah;  // reuse as [64 rows][16] merge buffer
#pragma unroll
  for (int i = 0; i < 8; ++i) {
#pragma unroll
    for (int t = 0; t < 8; ++t) {
      float h = s[i][0];
      float m = h;
      m = fminf(m, __shfl_xor(m, 1));
      m = fminf(m, __shfl_xor(m, 2));
      m = fminf(m, __shfl_xor(m, 4));
      m = fminf(m, __shfl_xor(m, 8));   // min within 16-lane half
      if (h == m) {                      // pop from owning lane(s)
#pragma unroll
        for (int j = 0; j < 7; ++j) s[i][j] = s[i][j + 1];
        s[i][7] = FLT_MAX;
      }
      if (l15 == 0) mbuf[(wm * 16 + i + 8 * hi) * 16 + wn * 8 + t] = m;
    }
  }
  __syncthreads();

  // ---- per row: merge the two waves' sorted 8-lists, sum 8 smallest ----
  if (tid < 64) {
    const float* a = &mbuf[tid * 16];
    const float* b = a + 8;
    int ia = 0, ib = 0;
    float sum = 0.0f;
#pragma unroll
    for (int t = 0; t < 8; ++t) {
      float va = a[ia], vb = b[ib];
      if (va <= vb) { sum += va; ++ia; } else { sum += vb; ++ib; }
    }
    atomicAdd(gsum, (double)sum);  // global_atomic_add_f64
  }
}

// ---------------------------------------------------------------------------
// Kernel 4: finalize  pseudo_entropy = mean(knn_dist^2) / ref_std
// ---------------------------------------------------------------------------
__global__ void finalize_kernel(const double* __restrict__ gsum,
                                const float* __restrict__ colsum,
                                const float* __restrict__ colsq,
                                float* __restrict__ out) {
  __shared__ float red[256];
  int t = threadIdx.x;
  float cs = colsum[t];
  float var = (colsq[t] - cs * cs * (1.0f / (float)N_PTS)) * (1.0f / (float)(N_PTS - 1));
  red[t] = var;
  __syncthreads();
  for (int ofs = 128; ofs > 0; ofs >>= 1) {
    if (t < ofs) red[t] += red[t + ofs];
    __syncthreads();
  }
  if (t == 0) {
    double ref_std = (double)red[0] / 256.0;
    out[0] = (float)(*gsum / ((double)N_PTS * (double)KNN) / ref_std);
  }
}

// ---------------------------------------------------------------------------
extern "C" void kernel_launch(void* const* d_in, const int* in_sizes, int n_in,
                              void* d_out, int out_size, void* d_ws, size_t ws_size,
                              hipStream_t stream) {
  (void)in_sizes; (void)n_in; (void)out_size; (void)ws_size;
  const float* e = (const float*)d_in[0];   // [8192, 256] fp32 (lp = d_in[1] unused)
  float* out = (float*)d_out;

  char* ws = (char*)d_ws;
  double* gsum  = (double*)ws;                  // 8 B
  float* colsum = (float*)(ws + 64);            // 256 floats
  float* colsq  = (float*)(ws + 64 + 1024);     // 256 floats
  float* sqn    = (float*)(ws + 64 + 2048);     // 8192 floats

  init_kernel     <<<1,           256, 0, stream>>>(gsum, colsum, colsq);
  rownorm_kernel  <<<N_PTS / 8,   256, 0, stream>>>(e, sqn);
  colstats_kernel <<<N_PTS / 256, 256, 0, stream>>>(e, colsum, colsq);
  gemm_topk_kernel<<<N_PTS / 64,  256, 0, stream>>>(e, sqn, gsum);
  finalize_kernel <<<1,           256, 0, stream>>>(gsum, colsum, colsq, out);
}
